// SequentialChart_6837587935431
// MI455X (gfx1250) — matrix operand order repaired and checked by reading, not compiled, not run on hardware
//
#include <hip/hip_runtime.h>

#define H        1024
#define BS       32
#define STEPS    128
#define AMB      32
#define START    256
#define CHART_LEN 384
#define TWO_H    2048
#define FIVE_H   5120
#define KPAD     2088   // LDS row stride (elements) -> conflict-friendly

typedef __bf16 bf16_t;
typedef __attribute__((ext_vector_type(4)))  __bf16 v4bf;
typedef __attribute__((ext_vector_type(8)))  __bf16 v8bf;
typedef __attribute__((ext_vector_type(16))) __bf16 v16bf;
typedef __attribute__((ext_vector_type(8)))  float  v8f;

__device__ __forceinline__ float sigmoidf_(float x) {
    return 1.0f / (1.0f + __expf(-x));
}

__device__ __forceinline__ v16bf load16_pair(const bf16_t* p0, const bf16_t* p1) {
    v8bf lo = *(const v8bf*)p0;
    v8bf hi = *(const v8bf*)p1;
    return __builtin_shufflevector(lo, hi, 0,1,2,3,4,5,6,7,8,9,10,11,12,13,14,15);
}

__device__ __forceinline__ v8f wmma_bf16(v16bf a, v16bf b, v8f c) {
    return __builtin_amdgcn_wmma_f32_16x16x32_bf16(false, a, false, b, (short)0, c, false, false);
}

// ---------------- U fp32 -> bf16 (runs once, U stays L2-resident) -------------
__global__ void convert_U_kernel(const float* __restrict__ U, bf16_t* __restrict__ Ubf) {
    int i = (blockIdx.x * 256 + threadIdx.x) * 4;
    float4 v = *(const float4*)(U + i);
    v4bf o = { (bf16_t)v.x, (bf16_t)v.y, (bf16_t)v.z, (bf16_t)v.w };
    *(v4bf*)(Ubf + i) = o;
}

// ---------------- ||energy_u|| -------------------------------------------------
__global__ void unorm_kernel(const float* __restrict__ eu, float* __restrict__ out) {
    __shared__ float red[256];
    int tid = threadIdx.x;
    float s = 0.f;
    for (int i = tid; i < H; i += 256) { float v = eu[i]; s += v * v; }
    red[tid] = s; __syncthreads();
    for (int off = 128; off > 0; off >>= 1) {
        if (tid < off) red[tid] += red[tid + off];
        __syncthreads();
    }
    if (tid == 0) out[0] = fmaxf(sqrtf(red[0]), 1e-8f);
}

// ---------------- main: one block per (step, batch) ---------------------------
__global__ __launch_bounds__(256)
void chart_step_kernel(const float* __restrict__ chart,
                       const int*   __restrict__ ops,
                       const float* __restrict__ bias,
                       const float* __restrict__ energy_u,
                       const bf16_t* __restrict__ Ubf,
                       const float* __restrict__ unorm_p,
                       float* __restrict__ out) {
    extern __shared__ char smem[];
    bf16_t* hhA    = (bf16_t*)smem;                 // [32][KPAD] gathered h (bf16)
    bf16_t* chS    = hhA + 32 * KPAD;               // [32][KPAD] c|h per amb (bf16)
    float*  red_hh = (float*)(chS + 32 * KPAD);     // [32]
    float*  red_he = red_hh + 32;                   // [32]
    float*  eS     = red_he + 32;                   // [32]
    float*  sS     = eS + 32;                       // [32]

    const int bid   = blockIdx.x;
    const int step  = bid >> 5;          // / BS
    const int batch = bid & 31;
    const int tid   = threadIdx.x;
    const int lane  = tid & 31;
    const int wave  = tid >> 5;
    const int l16   = lane & 15;
    const int hiG   = lane >> 4;         // 0/1
    const int hi8   = hiG * 8;
    const int hi16  = hiG * 16;

    const int* opsB = ops + ((size_t)step * BS + batch) * AMB * 2;

    if (tid < 32) { red_hh[tid] = 0.f; red_he[tid] = 0.f; }

    // ---- Phase A: gather h-halves of chart rows into LDS as bf16 ----
    // hh[a][0:1024] = chart[b, opL[a], H:2H] ; hh[a][1024:2048] = chart[b, opR[a], H:2H]
    for (int q = tid; q < 32 * (TWO_H / 4); q += 256) {
        int a    = q >> 9;               // 512 float4 per amb row
        int k4   = (q & 511) * 4;
        int side = k4 >> 10;             // 0 = left, 1 = right
        int col  = k4 & 1023;
        int src_row = opsB[a * 2 + side];
        const float* p = chart + (((size_t)batch * CHART_LEN + src_row) * TWO_H + H + col);
        float4 v = *(const float4*)p;
        v4bf o = { (bf16_t)v.x, (bf16_t)v.y, (bf16_t)v.z, (bf16_t)v.w };
        *(v4bf*)(hhA + a * KPAD + k4) = o;
    }
    __syncthreads();

    // ---- per-lane amb map + op indices for the cc gather ----
    int opl[16], opr[16];
    #pragma unroll
    for (int mt = 0; mt < 2; ++mt)
        #pragma unroll
        for (int r = 0; r < 8; ++r) {
            int li = mt * 8 + r;
            int m  = mt * 16 + hi8 + r;
            opl[li] = opsB[m * 2 + 0];
            opr[li] = opsB[m * 2 + 1];
        }

    float ss_hh[16], ss_he[16];
    #pragma unroll
    for (int i = 0; i < 16; ++i) { ss_hh[i] = 0.f; ss_he[i] = 0.f; }

    const int jw = wave * 128;           // this wave's j-slice base

    // ---- Phase B: GEMM (WMMA bf16) + gate epilogue, 8 j-tiles of 16 ----
    for (int jt = 0; jt < 8; ++jt) {
        const int jbase = jw + jt * 16;
        const int j     = jbase + l16;

        v8f acc[5][2];
        #pragma unroll
        for (int g = 0; g < 5; ++g)
            #pragma unroll
            for (int mt = 0; mt < 2; ++mt)
                acc[g][mt] = (v8f)(0.f);

        for (int kb = 0; kb < TWO_H; kb += 32) {
            // A tiles from LDS (row = amb), ISA split-K layout per lane half
            const bf16_t* a0p = hhA + (size_t)l16 * KPAD + kb + hi8;
            const bf16_t* a1p = hhA + (size_t)(16 + l16) * KPAD + kb + hi8;
            v16bf a0 = load16_pair(a0p, a0p + 16);
            v16bf a1 = load16_pair(a1p, a1p + 16);
            #pragma unroll
            for (int g = 0; g < 5; ++g) {
                // B[k][n] = U[n][k]; lane holds column n = jbase+l16, split-K layout per lane half
                const bf16_t* bp = Ubf + (size_t)(g * H + jbase + l16) * TWO_H + kb + hi8;
                v16bf bm = load16_pair(bp, bp + 16);
                acc[g][0] = wmma_bf16(a0, bm, acc[g][0]);
                acc[g][1] = wmma_bf16(a1, bm, acc[g][1]);
            }
        }

        // epilogue: gates, c/h, energy partials
        const float bi  = bias[0 * H + j];
        const float bfl = bias[1 * H + j];
        const float bfr = bias[2 * H + j];
        const float bo  = bias[3 * H + j];
        const float bu  = bias[4 * H + j];
        const float euj = energy_u[j];

        #pragma unroll
        for (int mt = 0; mt < 2; ++mt) {
            #pragma unroll
            for (int r = 0; r < 8; ++r) {
                const int li = mt * 8 + r;
                const int m  = mt * 16 + hi8 + r;
                float iv = sigmoidf_(acc[0][mt][r] + bi);
                float fl = sigmoidf_(acc[1][mt][r] + bfl);
                float fr = sigmoidf_(acc[2][mt][r] + bfr);
                float ov = sigmoidf_(acc[3][mt][r] + bo);
                float uv = tanhf(acc[4][mt][r] + bu);
                float ccl = chart[((size_t)batch * CHART_LEN + opl[li]) * TWO_H + j];
                float ccr = chart[((size_t)batch * CHART_LEN + opr[li]) * TWO_H + j];
                float c = fl * ccl + fr * ccr + iv * uv;
                float h = ov * tanhf(c);
                chS[(size_t)m * KPAD + j]     = (bf16_t)c;
                chS[(size_t)m * KPAD + H + j] = (bf16_t)h;
                ss_hh[li] += h * h;
                ss_he[li] += h * euj;
            }
        }
    }

    // ---- reduce energy partials across lanes/waves ----
    #pragma unroll
    for (int mt = 0; mt < 2; ++mt)
        #pragma unroll
        for (int r = 0; r < 8; ++r) {
            const int li = mt * 8 + r;
            const int m  = mt * 16 + hi8 + r;
            atomicAdd(&red_hh[m], ss_hh[li]);
            atomicAdd(&red_he[m], ss_he[li]);
        }
    __syncthreads();

    const float un = unorm_p[0];
    if (tid < 32) {
        float hn = fmaxf(sqrtf(red_hh[tid]), 1e-8f);
        eS[tid] = red_he[tid] / (un * hn);
    }
    __syncthreads();

    // softmax over amb (computed redundantly; weights published by tid<32)
    float mx = -1e30f;
    #pragma unroll
    for (int m = 0; m < 32; ++m) mx = fmaxf(mx, eS[m]);
    float sum = 0.f;
    #pragma unroll
    for (int m = 0; m < 32; ++m) sum += __expf(eS[m] - mx);
    if (tid < 32) sS[tid] = __expf(eS[tid] - mx) / sum;
    __syncthreads();

    float sreg[32];
    #pragma unroll
    for (int m = 0; m < 32; ++m) sreg[m] = sS[m];

    // ---- weighted combine -> chart row [batch, START+step, :] ----
    float* outRow = out + ((size_t)batch * CHART_LEN + START + step) * TWO_H;
    for (int t = tid; t < TWO_H; t += 256) {
        float a = 0.f;
        #pragma unroll
        for (int m = 0; m < 32; ++m)
            a += sreg[m] * (float)chS[(size_t)m * KPAD + t];
        outRow[t] = a;
    }
}

extern "C" void kernel_launch(void* const* d_in, const int* in_sizes, int n_in,
                              void* d_out, int out_size, void* d_ws, size_t ws_size,
                              hipStream_t stream) {
    const float* chart = (const float*)d_in[0];
    const int*   ops   = (const int*)d_in[1];
    // d_in[2] = start_index (== START, hardcoded)
    const float* U     = (const float*)d_in[3];
    const float* bias  = (const float*)d_in[4];
    const float* eu    = (const float*)d_in[5];
    float*       outp  = (float*)d_out;

    bf16_t* Ubf   = (bf16_t*)d_ws;
    float*  unorm = (float*)((char*)d_ws + (size_t)FIVE_H * TWO_H * sizeof(bf16_t));

    // untouched chart region passes through
    hipMemcpyAsync(d_out, d_in[0], (size_t)BS * CHART_LEN * TWO_H * sizeof(float),
                   hipMemcpyDeviceToDevice, stream);

    convert_U_kernel<<<(FIVE_H * TWO_H) / (256 * 4), 256, 0, stream>>>(U, Ubf);
    unorm_kernel<<<1, 256, 0, stream>>>(eu, unorm);

    const size_t smem_bytes = (size_t)2 * 32 * KPAD * sizeof(bf16_t) + 4 * 32 * sizeof(float);
    chart_step_kernel<<<dim3(STEPS * BS), 256, smem_bytes, stream>>>(
        chart, ops, bias, eu, Ubf, unorm, outp);
}
